// VectorQuantizer_25520695673418
// MI455X (gfx1250) — matrix operand order repaired and checked by reading, not compiled, not run on hardware
//
#include <hip/hip_runtime.h>
#include <hip/hip_bf16.h>

// ---------------- problem constants ----------------
constexpr int B = 16, D = 512, E = 256, K = 1024, T = 4096;
constexpr int N = B * T;                 // 65536 rows for quantization

// ---------------- workspace layout (bytes) ----------------
constexpr size_t OFF_XB    = 0;                                   // x as bf16 (B,D,T)
constexpr size_t OFF_WIN   = OFF_XB    + (size_t)B * D * T * 2;   // W_in  bf16 (E,D)
constexpr size_t OFF_WOUT  = OFF_WIN   + (size_t)E * D * 2;       // W_out bf16 (D,E)
constexpr size_t OFF_CNT   = OFF_WOUT  + (size_t)D * E * 2;       // normalized codebook^T bf16 (E,K)
constexpr size_t OFF_CB16  = OFF_CNT   + (size_t)E * K * 2;       // raw codebook bf16 (K,E)
constexpr size_t OFF_XPROJ = OFF_CB16  + (size_t)K * E * 2;       // x_proj f32 (B,E,T)
constexpr size_t OFF_XN    = OFF_XPROJ + (size_t)B * E * T * 4;   // xn bf16 (N,E)
constexpr size_t OFF_XQ    = OFF_XN    + (size_t)N * E * 2;       // quantized bf16 (B,E,T)
constexpr size_t OFF_LOSS  = OFF_XQ    + (size_t)B * E * T * 2;   // 16 f32 accumulators

// ---------------- output layout (elements of d_out) ----------------
constexpr size_t OUT_LOGITS = (size_t)B * D * T;                  // 33,554,432
constexpr size_t OUT_IDX    = OUT_LOGITS + (size_t)N * K;         // 100,663,296
constexpr size_t OUT_COMMIT = OUT_IDX + (size_t)B * T;
constexpr size_t OUT_CBL    = OUT_COMMIT + (size_t)B;

// ---------------- WMMA fragment types ----------------
typedef __attribute__((ext_vector_type(16))) __bf16 v16bf;
typedef __attribute__((ext_vector_type(8)))  __bf16 v8bf;
typedef __attribute__((ext_vector_type(4)))  __bf16 v4bf;
typedef __attribute__((ext_vector_type(8)))  float  v8f;

// A fragment: 16x32 bf16, M x K.  lane m = lane&15; lanes 0-15 hold K runs
// [0..7],[16..23]; lanes 16-31 hold [8..15],[24..31].  Two 16B loads.
__device__ __forceinline__ v16bf load_a_frag(const __bf16* __restrict__ tile,
                                             int ld, int lane) {
    const int m  = lane & 15;
    const int kb = (lane >> 4) << 3;          // 0 or 8
    const __bf16* r = tile + (size_t)m * ld + kb;
    v8bf lo = *(const v8bf*)(r);
    v8bf hi = *(const v8bf*)(r + 16);
    v16bf a;
#pragma unroll
    for (int i = 0; i < 8; ++i) { a[i] = lo[i]; a[i + 8] = hi[i]; }
    return a;
}

// B fragment: 32x16 bf16, K x N.  lane n = lane&15; lanes 0-15 hold K runs
// [0..7],[16..23]; lanes 16-31 hold [8..15],[24..31] (matches A).
__device__ __forceinline__ v16bf load_b_frag(const __bf16* __restrict__ tile,
                                             int ld, int lane) {
    const int n  = lane & 15;
    const int kb = (lane >> 4) << 3;
    v16bf b;
#pragma unroll
    for (int i = 0; i < 8; ++i) {
        b[i]     = tile[(size_t)(kb + i) * ld + n];
        b[i + 8] = tile[(size_t)(16 + kb + i) * ld + n];
    }
    return b;
}

__device__ __forceinline__ v8f wmma_bf16(v16bf a, v16bf b, v8f c) {
    return __builtin_amdgcn_wmma_f32_16x16x32_bf16(false, a, false, b,
                                                   (short)0, c, false, false);
}

// Software-pipelined 16x64 GEMM strip: C(16x64) += A(16xKD) * Bmat(KDx64),
// A row-major (lda), B row-major (ldb).  Double-buffers 1 A + 4 B fragments.
// The asm memory barrier pins the next-step loads ABOVE the current WMMA
// group so the scheduler cannot sink them back to their use point.
template <int KD>
__device__ __forceinline__ void gemm_strip_16x64(
    const __bf16* __restrict__ Abase, int lda,
    const __bf16* __restrict__ Bbase, int ldb,
    int lane, v8f acc[4]) {
    v16bf a_cur = load_a_frag(Abase, lda, lane);
    v16bf b_cur[4];
#pragma unroll
    for (int u = 0; u < 4; ++u) b_cur[u] = load_b_frag(Bbase + u * 16, ldb, lane);
#pragma unroll
    for (int kk = 0; kk < KD; kk += 32) {
        v16bf a_nxt, b_nxt[4];
        if (kk + 32 < KD) {                       // issue next-step loads first
            a_nxt = load_a_frag(Abase + kk + 32, lda, lane);
#pragma unroll
            for (int u = 0; u < 4; ++u)
                b_nxt[u] = load_b_frag(Bbase + (size_t)(kk + 32) * ldb + u * 16, ldb, lane);
        } else {
            a_nxt = a_cur;
#pragma unroll
            for (int u = 0; u < 4; ++u) b_nxt[u] = b_cur[u];
        }
        asm volatile("" ::: "memory");            // keep prefetch loads above WMMAs
#pragma unroll
        for (int u = 0; u < 4; ++u) acc[u] = wmma_bf16(a_cur, b_cur[u], acc[u]);
        a_cur = a_nxt;
#pragma unroll
        for (int u = 0; u < 4; ++u) b_cur[u] = b_nxt[u];
    }
}

// ---------------- kernel 1: weight-norm prep + codebook transforms ----------------
__global__ __launch_bounds__(256) void k_prep(
    const float* __restrict__ in_v,  const float* __restrict__ in_g,
    const float* __restrict__ out_v, const float* __restrict__ out_g,
    const float* __restrict__ cb,
    __bf16* __restrict__ Win, __bf16* __restrict__ Wout,
    __bf16* __restrict__ cnT, __bf16* __restrict__ cb16,
    float* __restrict__ lossAcc) {
    __shared__ float red[256];
    const int tid = threadIdx.x, bid = blockIdx.x;
    if (bid == 0 && tid < B) lossAcc[tid] = 0.0f;
    if (bid < E) {                                       // W_in row: norm over D=512
        float v0 = in_v[bid * D + tid];
        float v1 = in_v[bid * D + tid + 256];
        red[tid] = v0 * v0 + v1 * v1;
        __syncthreads();
        for (int s = 128; s > 0; s >>= 1) { if (tid < s) red[tid] += red[tid + s]; __syncthreads(); }
        float sc = in_g[bid] * rsqrtf(red[0]);
        Win[bid * D + tid]       = (__bf16)(v0 * sc);
        Win[bid * D + tid + 256] = (__bf16)(v1 * sc);
    } else if (bid < E + D) {                            // W_out row: norm over E=256
        int d = bid - E;
        float v = out_v[d * E + tid];
        red[tid] = v * v;
        __syncthreads();
        for (int s = 128; s > 0; s >>= 1) { if (tid < s) red[tid] += red[tid + s]; __syncthreads(); }
        Wout[d * E + tid] = (__bf16)(v * out_g[d] * rsqrtf(red[0]));
    } else {                                             // codebook row k: normalize + transpose
        int k = bid - E - D;
        float v = cb[k * E + tid];
        red[tid] = v * v;
        __syncthreads();
        for (int s = 128; s > 0; s >>= 1) { if (tid < s) red[tid] += red[tid + s]; __syncthreads(); }
        float inv = 1.0f / fmaxf(sqrtf(red[0]), 1e-12f);
        cnT[(size_t)tid * K + k] = (__bf16)(v * inv);    // (E,K) for B fragments
        cb16[(size_t)k * E + tid] = (__bf16)v;           // raw bf16 for gather
    }
}

// ---------------- kernel 2: x f32 -> bf16 ----------------
__global__ __launch_bounds__(256) void k_cvt_x(const float* __restrict__ x,
                                               __bf16* __restrict__ xb) {
    size_t i = ((size_t)blockIdx.x * 256 + threadIdx.x) * 4;
    float4 v = *(const float4*)(x + i);
    v4bf o;
    o[0] = (__bf16)v.x; o[1] = (__bf16)v.y; o[2] = (__bf16)v.z; o[3] = (__bf16)v.w;
    *(v4bf*)(xb + i) = o;
}

// ---------------- kernel 3: x_proj = W_in @ x + in_b  (WMMA, 16x64 per wave) ----------------
__global__ __launch_bounds__(128) void k_in_proj(
    const __bf16* __restrict__ xb, const __bf16* __restrict__ Win,
    const float* __restrict__ in_b, float* __restrict__ xproj) {
    const int lane = threadIdx.x & 31, wave = threadIdx.x >> 5;
    const int b  = blockIdx.z;
    const int e0 = blockIdx.y * 16;
    const int t0 = blockIdx.x * 256 + wave * 64;
    v8f acc[4] = {{}, {}, {}, {}};
    gemm_strip_16x64<D>(Win + (size_t)e0 * D, D,
                        xb + (size_t)b * D * T + t0, T, lane, acc);
    const int col = lane & 15, ro = (lane >> 4) << 3;
    float* outp = xproj + (size_t)b * E * T + (size_t)(e0 + ro) * T + t0 + col;
#pragma unroll
    for (int r = 0; r < 8; ++r) {
        float bias = in_b[e0 + ro + r];
#pragma unroll
        for (int u = 0; u < 4; ++u)
            outp[(size_t)r * T + u * 16] = acc[u][r] + bias;
    }
}

// ---------------- kernel 4: row L2-normalize + transpose to (N,E) bf16 ----------------
__global__ __launch_bounds__(256) void k_rownorm(const float* __restrict__ xproj,
                                                 __bf16* __restrict__ xn) {
    const int tid = threadIdx.x;
    const int lt = tid & 31;          // t within 32-wide tile
    const int le = tid >> 5;          // e phase 0..7
    const int b  = blockIdx.y;
    const int t0 = blockIdx.x * 32;
    const float* base = xproj + (size_t)b * E * T + t0 + lt;
    float vals[32];
    float s = 0.0f;
#pragma unroll
    for (int j = 0; j < 32; ++j) {                     // e = le + 8*j, coalesced over lt
        float v = base[(size_t)(le + 8 * j) * T];
        vals[j] = v; s += v * v;
    }
    __shared__ float red[256];
    __shared__ float inv[32];
    red[tid] = s;
    __syncthreads();
    if (le == 0) {
        float tot = 0.0f;
#pragma unroll
        for (int p = 0; p < 8; ++p) tot += red[p * 32 + lt];
        inv[lt] = 1.0f / fmaxf(sqrtf(tot), 1e-12f);
    }
    __syncthreads();
    const float sc = inv[lt];
    __shared__ __bf16 tile[32][64];
    const int row_base = b * T + t0;
    for (int e0 = 0; e0 < E; e0 += 64) {
#pragma unroll
        for (int jj = 0; jj < 8; ++jj) {
            int j = e0 / 8 + jj;
            int e = le + 8 * j;
            tile[lt][e - e0] = (__bf16)(vals[j] * sc);
        }
        __syncthreads();
#pragma unroll
        for (int p = 0; p < 8; ++p) {                  // coalesced (row, e) writes
            int r  = p * 4 + (tid >> 6);
            int ei = tid & 63;
            xn[(size_t)(row_base + r) * E + e0 + ei] = tile[r][ei];
        }
        __syncthreads();
    }
}

// ---------------- kernel 5: logits = 2*(xn @ cn^T) - 2  (WMMA, 16x64 per wave) ----------------
__global__ __launch_bounds__(128) void k_dist(
    const __bf16* __restrict__ xn, const __bf16* __restrict__ cnT,
    float* __restrict__ logits) {
    const int lane = threadIdx.x & 31, wave = threadIdx.x >> 5;
    const int n0 = blockIdx.y * 16;
    const int c0 = blockIdx.x * 256 + wave * 64;
    v8f acc[4] = {{}, {}, {}, {}};
    gemm_strip_16x64<E>(xn + (size_t)n0 * E, E, cnT + c0, K, lane, acc);
    const int col = lane & 15, ro = (lane >> 4) << 3;
    float* outp = logits + (size_t)(n0 + ro) * K + c0 + col;
#pragma unroll
    for (int r = 0; r < 8; ++r) {                      // -dist, since ||xn||=||cn||=1
#pragma unroll
        for (int u = 0; u < 4; ++u)
            outp[(size_t)r * K + u * 16] = 2.0f * acc[u][r] - 2.0f;
    }
}

// ---------------- kernel 6: argmax over K per row ----------------
__global__ __launch_bounds__(256) void k_argmax(const float* __restrict__ logits,
                                                int* __restrict__ idxs) {
    const int row = blockIdx.x, tid = threadIdx.x;
    const float* p = logits + (size_t)row * K;
    float bv = -3.402823466e38f; int bi = 0;
#pragma unroll
    for (int j = 0; j < 4; ++j) {
        int i = tid + j * 256;                         // increasing index -> first-max kept
        float v = p[i];
        if (v > bv) { bv = v; bi = i; }
    }
    __shared__ float sv[256];
    __shared__ int   si[256];
    sv[tid] = bv; si[tid] = bi;
    __syncthreads();
    for (int s = 128; s > 0; s >>= 1) {
        if (tid < s) {
            if (sv[tid + s] > sv[tid] ||
                (sv[tid + s] == sv[tid] && si[tid + s] < si[tid])) {
                sv[tid] = sv[tid + s]; si[tid] = si[tid + s];
            }
        }
        __syncthreads();
    }
    if (tid == 0) idxs[row] = si[0];
}

// ---------------- kernel 7: gather xq[b,e,t] = cb16[idx[b,t], e] ----------------
__global__ __launch_bounds__(256) void k_gather(const int* __restrict__ idxs,
                                                const __bf16* __restrict__ cb16,
                                                __bf16* __restrict__ xq) {
    size_t i = (size_t)blockIdx.x * 256 + threadIdx.x;   // over B*E*T
    int t = (int)(i & (T - 1));
    int e = (int)((i >> 12) & (E - 1));
    int b = (int)(i >> 20);                              // E*T = 2^20
    int idx = idxs[b * T + t];
    xq[i] = cb16[(size_t)idx * E + e];                   // coalesced store, L2-resident gather
}

// ---------------- kernel 8: out = W_out @ xq + out_b  (WMMA, 16x64 per wave) ----------------
__global__ __launch_bounds__(128) void k_out_proj(
    const __bf16* __restrict__ xq, const __bf16* __restrict__ Wout,
    const float* __restrict__ out_b, float* __restrict__ out) {
    const int lane = threadIdx.x & 31, wave = threadIdx.x >> 5;
    const int b  = blockIdx.z;
    const int d0 = blockIdx.y * 16;
    const int t0 = blockIdx.x * 256 + wave * 64;
    v8f acc[4] = {{}, {}, {}, {}};
    gemm_strip_16x64<E>(Wout + (size_t)d0 * E, E,
                        xq + (size_t)b * E * T + t0, T, lane, acc);
    const int col = lane & 15, ro = (lane >> 4) << 3;
    float* outp = out + (size_t)b * D * T + (size_t)(d0 + ro) * T + t0 + col;
#pragma unroll
    for (int r = 0; r < 8; ++r) {
        float bias = out_b[d0 + ro + r];
#pragma unroll
        for (int u = 0; u < 4; ++u)
            outp[(size_t)r * T + u * 16] = acc[u][r] + bias;
    }
}

// ---------------- kernel 9: per-batch sum of (x_proj - codebook[idx])^2 ----------------
__global__ __launch_bounds__(256) void k_loss(const float* __restrict__ xproj,
                                              const float* __restrict__ cb,
                                              const int* __restrict__ idxs,
                                              float* __restrict__ acc) {
    const int b = blockIdx.y;
    size_t i = (size_t)blockIdx.x * 256 + threadIdx.x;   // within b: [0, E*T)
    int t = (int)(i & (T - 1));
    int e = (int)(i >> 12);
    int idx = idxs[b * T + t];
    float d = xproj[(size_t)b * E * T + i] - cb[(size_t)idx * E + e];
    __shared__ float red[256];
    red[threadIdx.x] = d * d;
    __syncthreads();
    for (int s = 128; s > 0; s >>= 1) { if (threadIdx.x < s) red[threadIdx.x] += red[threadIdx.x + s]; __syncthreads(); }
    if (threadIdx.x == 0) atomicAdd(acc + b, red[0]);
}

// ---------------- kernel 10: finalize losses ----------------
__global__ void k_finalize(const float* __restrict__ acc,
                           float* __restrict__ commit, float* __restrict__ cbl) {
    int b = threadIdx.x;
    if (b < B) {
        float v = acc[b] / (float)(E * T);
        commit[b] = v;
        cbl[b]    = v;    // codebook loss forward value is identical
    }
}

extern "C" void kernel_launch(void* const* d_in, const int* in_sizes, int n_in,
                              void* d_out, int out_size, void* d_ws, size_t ws_size,
                              hipStream_t stream) {
    (void)in_sizes; (void)n_in; (void)out_size; (void)ws_size;
    const float* x     = (const float*)d_in[0];
    const float* in_v  = (const float*)d_in[1];
    const float* in_g  = (const float*)d_in[2];
    const float* in_b  = (const float*)d_in[3];
    const float* out_v = (const float*)d_in[4];
    const float* out_g = (const float*)d_in[5];
    const float* out_b = (const float*)d_in[6];
    const float* cb    = (const float*)d_in[7];

    char* w = (char*)d_ws;
    __bf16* xb      = (__bf16*)(w + OFF_XB);
    __bf16* Win     = (__bf16*)(w + OFF_WIN);
    __bf16* Wout    = (__bf16*)(w + OFF_WOUT);
    __bf16* cnT     = (__bf16*)(w + OFF_CNT);
    __bf16* cb16    = (__bf16*)(w + OFF_CB16);
    float*  xproj   = (float*) (w + OFF_XPROJ);
    __bf16* xn      = (__bf16*)(w + OFF_XN);
    __bf16* xq      = (__bf16*)(w + OFF_XQ);
    float*  lossAcc = (float*) (w + OFF_LOSS);

    float* out    = (float*)d_out;
    float* logits = out + OUT_LOGITS;
    int*   idxs   = (int*)(out + OUT_IDX);
    float* commit = out + OUT_COMMIT;
    float* cbl    = out + OUT_CBL;

    k_prep<<<E + D + K, 256, 0, stream>>>(in_v, in_g, out_v, out_g, cb,
                                          Win, Wout, cnT, cb16, lossAcc);
    k_cvt_x<<<(B * D * T) / (256 * 4), 256, 0, stream>>>(x, xb);
    k_in_proj<<<dim3(T / 256, E / 16, B), 128, 0, stream>>>(xb, Win, in_b, xproj);
    k_rownorm<<<dim3(T / 32, B), 256, 0, stream>>>(xproj, xn);
    k_dist<<<dim3(K / 256, N / 16), 128, 0, stream>>>(xn, cnT, logits);
    k_argmax<<<N, 256, 0, stream>>>(logits, idxs);
    k_gather<<<(B * E * T) / 256, 256, 0, stream>>>(idxs, cb16, xq);
    k_out_proj<<<dim3(T / 256, D / 16, B), 128, 0, stream>>>(xq, Wout, out_b, out);
    k_loss<<<dim3((E * T) / 256, B), 256, 0, stream>>>(xproj, cb, idxs, lossAcc);
    k_finalize<<<1, 32, 0, stream>>>(lossAcc, commit, cbl);
}
